// KDEAttention_83202106458478
// MI455X (gfx1250) — matrix operand order repaired and checked by reading, not compiled, not run on hardware
//
#include <hip/hip_runtime.h>

typedef _Float16 v16h __attribute__((ext_vector_type(16)));
typedef _Float16 v8h  __attribute__((ext_vector_type(8)));
typedef float    v8f  __attribute__((ext_vector_type(8)));
typedef float    v4f  __attribute__((ext_vector_type(4)));

#define CH     256
#define RED    64          // CH / 4
#define NPIX   4096        // 64*64
#define NBATCH 32
#define BT     64          // block tile (rows == cols)
#define LDSP   72          // padded LDS row stride in halves (144B = 9*16B, conflict-free)
#define GAMMA  0.5f

// XOR-butterfly add across lanes via ds_swizzle (group-of-32: and=0x1f, or=0, xor=m)
#define XOR_ADD(v, m) \
  ((v) + __int_as_float(__builtin_amdgcn_ds_swizzle(__float_as_int(v), ((m) << 10) | 0x1f)))

// ---------------------------------------------------------------------------
// Kernel 1: sq[b*CH+c] = sum_n x[b,c,n]^2
// ---------------------------------------------------------------------------
__global__ __launch_bounds__(256) void sq_kernel(const float* __restrict__ x,
                                                 float* __restrict__ sq) {
  const int row  = blockIdx.x * 8 + (threadIdx.x >> 5);   // 0 .. 8191
  const int lane = threadIdx.x & 31;
  const float4* p = (const float4*)(x + (size_t)row * NPIX);
  float s = 0.f;
  for (int i = lane; i < NPIX / 4; i += 32) {
    float4 v = p[i];
    s += v.x * v.x + v.y * v.y + v.z * v.z + v.w * v.w;
  }
  s = XOR_ADD(s, 16);
  s = XOR_ADD(s, 8);
  s = XOR_ADD(s, 4);
  s = XOR_ADD(s, 2);
  s = XOR_ADD(s, 1);
  if (lane == 0) sq[row] = s;
}

// ---------------------------------------------------------------------------
// helpers for the gram kernel
// ---------------------------------------------------------------------------
__device__ __forceinline__ v8h pack8(float4 u, float4 v) {
  v8h r = {(_Float16)u.x, (_Float16)u.y, (_Float16)u.z, (_Float16)u.w,
           (_Float16)v.x, (_Float16)v.y, (_Float16)v.z, (_Float16)v.w};
  return r;
}

__device__ __forceinline__ v16h frag16(const _Float16* lo, const _Float16* hi) {
  v8h a = *(const v8h*)lo;
  v8h b = *(const v8h*)hi;
  return __builtin_shufflevector(a, b, 0, 1, 2, 3, 4, 5, 6, 7,
                                 8, 9, 10, 11, 12, 13, 14, 15);
}

// ---------------------------------------------------------------------------
// Kernel 2: fused Gram (f16 WMMA, fp32 accum) + gaussian kernel + row-sum.
// Grid (16, NBATCH), block 128 = 4 waves.  Each wave owns a 2x2 group of
// 16x16 subtiles (4 accumulators, shared A/B fragments).  LDS double-buffered:
// next panel's global loads are issued before this panel's WMMAs.
// ---------------------------------------------------------------------------
__global__ __launch_bounds__(128) void gram_kde_kernel(const float* __restrict__ x,
                                                       const float* __restrict__ sq,
                                                       float* __restrict__ density) {
  __shared__ _Float16 As[2][BT][LDSP];
  __shared__ _Float16 Bs[2][BT][LDSP];

  const int b       = blockIdx.y;
  const int rowBase = (blockIdx.x >> 2) * BT;
  const int colBase = (blockIdx.x & 3) * BT;
  const int t    = threadIdx.x;
  const int wid  = t >> 5;
  const int lane = t & 31;
  const int sel  = lane >> 4;     // half-wave select
  const int lRow = lane & 15;     // row/col within 16-group

  // 2x2 subtile group per wave over the 4x4 subtile grid
  const int rT0 = (wid >> 1) * 2, rT1 = rT0 + 1;
  const int cT0 = (wid & 1) * 2,  cT1 = cT0 + 1;

  const float* xb = x + (size_t)b * ((size_t)CH * NPIX);

  // panel staging: thread t loads 32 floats of row (t>>1), k-chunk (t&1)*32,
  // for both the A (row) panel and the B (col) panel.
  const int pRow = t >> 1;
  const int pK   = (t & 1) * 32;
  const float* aSrc = xb + (size_t)(rowBase + pRow) * NPIX + pK;
  const float* bSrc = xb + (size_t)(colBase + pRow) * NPIX + pK;

  v8f acc00 = {}, acc01 = {}, acc10 = {}, acc11 = {};

  float4 ra[8], rb[8];
  { // prologue: stage panel kt=0
    const float4* a4 = (const float4*)aSrc;
    const float4* b4 = (const float4*)bSrc;
#pragma unroll
    for (int i = 0; i < 8; ++i) { ra[i] = a4[i]; rb[i] = b4[i]; }
#pragma unroll
    for (int i = 0; i < 4; ++i) {
      *(v8h*)&As[0][pRow][pK + i * 8] = pack8(ra[2 * i], ra[2 * i + 1]);
      *(v8h*)&Bs[0][pRow][pK + i * 8] = pack8(rb[2 * i], rb[2 * i + 1]);
    }
  }

  int cur = 0;
  for (int kt = 0; kt < NPIX; kt += BT) {
    __syncthreads();
    const bool hasNext = (kt + BT) < NPIX;
    if (hasNext) {  // issue next panel's global loads before compute
      const float4* a4 = (const float4*)(aSrc + kt + BT);
      const float4* b4 = (const float4*)(bSrc + kt + BT);
#pragma unroll
      for (int i = 0; i < 8; ++i) { ra[i] = a4[i]; rb[i] = b4[i]; }
    }

#pragma unroll
    for (int ks = 0; ks < BT; ks += 32) {
      // A fragments (16x32): per-lane chunks [ks+sel*8 .. +7], [ks+16+sel*8 .. +7]
      const _Float16* a0 = &As[cur][rT0 * 16 + lRow][ks + sel * 8];
      const _Float16* a1 = &As[cur][rT1 * 16 + lRow][ks + sel * 8];
      v16h aF0 = frag16(a0, a0 + 16);
      v16h aF1 = frag16(a1, a1 + 16);
      // B fragments (32x16): per-lane chunks [ks+sel*8 .. +7], [ks+16+sel*8 .. +7]
      const _Float16* b0 = &Bs[cur][cT0 * 16 + lRow][ks + sel * 8];
      const _Float16* b1 = &Bs[cur][cT1 * 16 + lRow][ks + sel * 8];
      v16h bF0 = frag16(b0, b0 + 16);
      v16h bF1 = frag16(b1, b1 + 16);

      acc00 = __builtin_amdgcn_wmma_f32_16x16x32_f16(false, aF0, false, bF0,
                                                     (short)0, acc00, false, false);
      acc01 = __builtin_amdgcn_wmma_f32_16x16x32_f16(false, aF0, false, bF1,
                                                     (short)0, acc01, false, false);
      acc10 = __builtin_amdgcn_wmma_f32_16x16x32_f16(false, aF1, false, bF0,
                                                     (short)0, acc10, false, false);
      acc11 = __builtin_amdgcn_wmma_f32_16x16x32_f16(false, aF1, false, bF1,
                                                     (short)0, acc11, false, false);
    }

    if (hasNext) {  // convert + store next panel into the other buffer
#pragma unroll
      for (int i = 0; i < 4; ++i) {
        *(v8h*)&As[1 - cur][pRow][pK + i * 8] = pack8(ra[2 * i], ra[2 * i + 1]);
        *(v8h*)&Bs[1 - cur][pRow][pK + i * 8] = pack8(rb[2 * i], rb[2 * i + 1]);
      }
    }
    cur ^= 1;
  }

  // Epilogue. D layout: VGPR v, lane -> (M = v + 8*sel, N = lRow).
  const int col0 = colBase + cT0 * 16 + lRow;
  const int col1 = colBase + cT1 * 16 + lRow;
  const float* sqb = sq + b * CH;
  const float sqc0 = sqb[col0];
  const float sqc1 = sqb[col1];
  float* dens = density + b * CH;

#pragma unroll
  for (int v = 0; v < 8; ++v) {
    {
      const int row = rowBase + rT0 * 16 + sel * 8 + v;
      const float sqr = sqb[row];
      float d20 = fmaxf(sqr + sqc0 - 2.f * acc00[v], 0.f);
      float d21 = fmaxf(sqr + sqc1 - 2.f * acc01[v], 0.f);
      float kv = ((row == col0) ? 1.f : __expf(-0.5f * d20)) +
                 ((row == col1) ? 1.f : __expf(-0.5f * d21));
      kv = XOR_ADD(kv, 1);
      kv = XOR_ADD(kv, 2);
      kv = XOR_ADD(kv, 4);
      kv = XOR_ADD(kv, 8);
      if (lRow == 0) atomicAdd(&dens[row], kv);
    }
    {
      const int row = rowBase + rT1 * 16 + sel * 8 + v;
      const float sqr = sqb[row];
      float d20 = fmaxf(sqr + sqc0 - 2.f * acc10[v], 0.f);
      float d21 = fmaxf(sqr + sqc1 - 2.f * acc11[v], 0.f);
      float kv = ((row == col0) ? 1.f : __expf(-0.5f * d20)) +
                 ((row == col1) ? 1.f : __expf(-0.5f * d21));
      kv = XOR_ADD(kv, 1);
      kv = XOR_ADD(kv, 2);
      kv = XOR_ADD(kv, 4);
      kv = XOR_ADD(kv, 8);
      if (lRow == 0) atomicAdd(&dens[row], kv);
    }
  }
}

// ---------------------------------------------------------------------------
// Kernel 3: tiny MLP: scale[b,c] = sigmoid(relu(dens/CH @ w1 + b1) @ w2 + b2) + GAMMA
// ---------------------------------------------------------------------------
__global__ __launch_bounds__(256) void mlp_kernel(const float* __restrict__ density,
                                                  const float* __restrict__ w1,
                                                  const float* __restrict__ b1,
                                                  const float* __restrict__ w2,
                                                  const float* __restrict__ b2,
                                                  float* __restrict__ scale) {
  __shared__ float dens[CH];
  __shared__ float hid[RED];
  const int b = blockIdx.x;
  const int t = threadIdx.x;
  dens[t] = density[b * CH + t] * (1.0f / CH);
  __syncthreads();
  if (t < RED) {
    float s = b1[t];
    for (int j = 0; j < CH; ++j) s += dens[j] * w1[j * RED + t];
    hid[t] = fmaxf(s, 0.f);
  }
  __syncthreads();
  float s = b2[t];
  for (int j = 0; j < RED; ++j) s += hid[j] * w2[j * CH + t];
  scale[b * CH + t] = 1.f / (1.f + __expf(-s)) + GAMMA;
}

// ---------------------------------------------------------------------------
// Kernel 4: out = x * scale[b,c].  Last use of x -> NT load; streaming NT store.
// Uses native clang vector type (v4f) so the nontemporal builtins accept it.
// ---------------------------------------------------------------------------
__global__ __launch_bounds__(256) void scale_kernel(const float* __restrict__ x,
                                                    const float* __restrict__ scale,
                                                    float* __restrict__ out) {
  const size_t i = (size_t)blockIdx.x * blockDim.x + threadIdx.x;  // float4 index
  const int bc = (int)(i >> 10);            // NPIX/4 = 1024 float4 per (b,c)
  const float s = scale[bc];
  v4f v = __builtin_nontemporal_load((const v4f*)x + i);
  v *= s;
  __builtin_nontemporal_store(v, (v4f*)out + i);
}

// ---------------------------------------------------------------------------
extern "C" void kernel_launch(void* const* d_in, const int* in_sizes, int n_in,
                              void* d_out, int out_size, void* d_ws, size_t ws_size,
                              hipStream_t stream) {
  const float* x  = (const float*)d_in[0];
  const float* w1 = (const float*)d_in[1];
  const float* b1 = (const float*)d_in[2];
  const float* w2 = (const float*)d_in[3];
  const float* b2 = (const float*)d_in[4];
  float* out = (float*)d_out;

  float* sq      = (float*)d_ws;                 // 8192 floats
  float* density = sq + NBATCH * CH;             // 8192 floats
  float* scale   = density + NBATCH * CH;        // 8192 floats

  (void)hipMemsetAsync(density, 0, (size_t)NBATCH * CH * sizeof(float), stream);

  sq_kernel<<<NBATCH * CH / 8, 256, 0, stream>>>(x, sq);

  gram_kde_kernel<<<dim3(16, NBATCH), 128, 0, stream>>>(x, sq, density);

  mlp_kernel<<<NBATCH, 256, 0, stream>>>(density, w1, b1, w2, b2, scale);

  const size_t nv4 = (size_t)NBATCH * CH * NPIX / 4;
  scale_kernel<<<(unsigned)(nv4 / 256), 256, 0, stream>>>(x, scale, out);
}
